// DySample_35845797052916
// MI455X (gfx1250) — matrix operand order repaired and checked by reading, not compiled, not run on hardware
//
#include <hip/hip_runtime.h>
#include <hip/hip_bf16.h>

typedef __attribute__((ext_vector_type(16))) _Float16 v16h;
typedef __attribute__((ext_vector_type(8)))  _Float16 v8h;
typedef __attribute__((ext_vector_type(8)))  float    v8f;

#define WMMA_F16(a,b,c) __builtin_amdgcn_wmma_f32_16x16x32_f16(false,(a),false,(b),(short)0,(c),false,false)

// Problem constants (from reference)
constexpr int B_  = 8;
constexpr int C_  = 128;
constexpr int H_  = 128;
constexpr int W_  = 128;
constexpr int HW_ = H_ * W_;
constexpr int OH_ = 256;
constexpr int OW_ = 256;
constexpr int OHW_ = OH_ * OW_;
constexpr int CO_ = 32;   // offset branch channels (2*GROUPS*s^2)
constexpr int CP_ = 64;   // pointwise output channels

// ---------------------------------------------------------------------------
// K0: zero the BN stats accumulator (sum[128], sumsq[128])
__global__ void k0_zero(float* __restrict__ stats) {
    stats[threadIdx.x] = 0.0f;   // 256 threads
}

// ---------------------------------------------------------------------------
// K1: 1x1 conv 128->32 as WMMA GEMM.  off1[b,o,hw] = sum_c w1[o,c]*x[b,c,hw] + b1[o]
// Block: 256 thr (8 waves). Each wave computes one 32x16 tile (2 M-tiles).
__global__ __launch_bounds__(256) void k1_conv1x1(
    const float* __restrict__ x, const float* __restrict__ w1,
    const float* __restrict__ b1, float* __restrict__ off1)
{
    __shared__ _Float16 wl[CO_ * C_];            // 32x128 f16 weights
    const int tid = threadIdx.x;
    for (int i = tid; i < CO_ * C_; i += 256) wl[i] = (_Float16)w1[i];
    __syncthreads();

    const int wave = tid >> 5, lane = tid & 31;
    const int g = lane >> 4, ln = lane & 15;     // lane group / lane-in-group
    const int b = blockIdx.y;
    const int hw0 = blockIdx.x * 128 + wave * 16;
    const float* xb = x + (size_t)b * C_ * HW_;

    v8f acc0 = {}; v8f acc1 = {};
    for (int kk = 0; kk < C_; kk += 32) {
        // B fragment: K = kk + g*16 + j, N = hw0 + ln (coalesced across lanes)
        v16h bf;
        const int kb = kk + g * 16;
        #pragma unroll
        for (int j = 0; j < 16; ++j)
            bf[j] = (_Float16)xb[(size_t)(kb + j) * HW_ + hw0 + ln];
        // A fragments from LDS: rows m, K chunks [g*16..g*16+7] and [g*16+8..+7]
        #pragma unroll
        for (int mt = 0; mt < 2; ++mt) {
            const int m = mt * 16 + ln;
            const int base = m * C_ + kk + g * 16;
            v8h lo = *(const v8h*)&wl[base];
            v8h hi = *(const v8h*)&wl[base + 8];
            v16h af;
            #pragma unroll
            for (int j = 0; j < 8; ++j) { af[j] = lo[j]; af[j + 8] = hi[j]; }
            if (mt == 0) acc0 = WMMA_F16(af, bf, acc0);
            else         acc1 = WMMA_F16(af, bf, acc1);
        }
    }
    #pragma unroll
    for (int v = 0; v < 8; ++v) {
        const int m = v + g * 8;
        off1[(size_t)(b * CO_ + m)      * HW_ + hw0 + ln] = acc0[v] + b1[m];
        off1[(size_t)(b * CO_ + m + 16) * HW_ + hw0 + ln] = acc1[v] + b1[m + 16];
    }
}

// ---------------------------------------------------------------------------
// K2: 3x3 dilated(2) conv 32->32 (pad 2, zero) as 9 tap-GEMMs (K=32 each),
// fused epilogue: offset = conv/4 + init_pos (init_pos = +-0.25 by channel parity).
__global__ __launch_bounds__(256) void k2_conv3x3d2(
    const float* __restrict__ off1, const float* __restrict__ w2,
    float* __restrict__ off)
{
    __shared__ _Float16 wl[9 * CO_ * CO_];       // [tap][o][i] f16 (18 KB)
    const int tid = threadIdx.x;
    for (int i = tid; i < 9 * CO_ * CO_; i += 256) {
        const int tap = i >> 10, o = (i >> 5) & 31, ii = i & 31;
        wl[i] = (_Float16)w2[(o * CO_ + ii) * 9 + tap];
    }
    __syncthreads();

    const int wave = tid >> 5, lane = tid & 31;
    const int g = lane >> 4, ln = lane & 15;
    const int b = blockIdx.y;
    const int hw0 = blockIdx.x * 128 + wave * 16;
    const int h = hw0 >> 7, w0 = hw0 & 127;      // tile lies within one row
    const float* ob = off1 + (size_t)b * CO_ * HW_;

    v8f acc0 = {}; v8f acc1 = {};
    for (int tap = 0; tap < 9; ++tap) {
        const int dy = (tap / 3 - 1) * 2, dx = (tap % 3 - 1) * 2;
        const int hh = h + dy;
        if (hh < 0 || hh >= H_) continue;        // wave-uniform skip (zero pad)
        const int ww = w0 + ln + dx;
        const bool inb = (ww >= 0) && (ww < W_);
        const int wwc = inb ? ww : 0;
        v16h bf;
        const int kb = g * 16;
        #pragma unroll
        for (int j = 0; j < 16; ++j) {
            const float v = ob[(size_t)(kb + j) * HW_ + hh * W_ + wwc];
            bf[j] = (_Float16)(inb ? v : 0.0f);
        }
        #pragma unroll
        for (int mt = 0; mt < 2; ++mt) {
            const int m = mt * 16 + ln;
            const int base = (tap * CO_ + m) * CO_ + g * 16;
            v8h lo = *(const v8h*)&wl[base];
            v8h hi = *(const v8h*)&wl[base + 8];
            v16h af;
            #pragma unroll
            for (int j = 0; j < 8; ++j) { af[j] = lo[j]; af[j + 8] = hi[j]; }
            if (mt == 0) acc0 = WMMA_F16(af, bf, acc0);
            else         acc1 = WMMA_F16(af, bf, acc1);
        }
    }
    #pragma unroll
    for (int v = 0; v < 8; ++v) {
        { const int m = v + g * 8;               // channels 0..15: parity of bit0
          const float ip = (m & 1) ? 0.25f : -0.25f;
          off[(size_t)(b * CO_ + m) * HW_ + hw0 + ln] = acc0[v] * 0.25f + ip; }
        { const int m = v + g * 8 + 16;          // channels 16..31: parity of bit1
          const float ip = ((m >> 1) & 1) ? 0.25f : -0.25f;
          off[(size_t)(b * CO_ + m) * HW_ + hw0 + ln] = acc1[v] * 0.25f + ip; }
    }
}

// ---------------------------------------------------------------------------
// K3: DySample bilinear grid_sample (border clamp) -> up[b, c, 256, 256]
// One thread per (b,g,oh,ow); 32 channels of the group share coordinates.
__global__ __launch_bounds__(256) void k3_sample(
    const float* __restrict__ x, const float* __restrict__ off,
    float* __restrict__ up)
{
    const int ow = threadIdx.x;
    const int oh = blockIdx.x;
    const int bg = blockIdx.y;
    const int b = bg >> 2, g = bg & 3;
    const int h = oh >> 1, w = ow >> 1;
    const int k = g * 4 + (oh & 1) * 2 + (ow & 1);   // pixel-shuffle decode

    const float* ofb = off + (size_t)b * CO_ * HW_;
    const float offx = ofb[(size_t)k        * HW_ + h * W_ + w];
    const float offy = ofb[(size_t)(k + 16) * HW_ + h * W_ + w];

    const float PI = 3.14159265358979f;
    const float cwv = (float)w + __sinf(PI * (float)(w + 1) * (1.0f / 128.0f));
    const float chv = (float)h + __sinf(PI * (float)(h + 1) * (1.0f / 128.0f));
    float ix = fminf(fmaxf(cwv + offx - 0.5f, 0.0f), 127.0f);
    float iy = fminf(fmaxf(chv + offy - 0.5f, 0.0f), 127.0f);
    const float x0f = floorf(ix), y0f = floorf(iy);
    const float wx = ix - x0f, wy = iy - y0f;
    const int x0 = (int)x0f, y0 = (int)y0f;
    const int x1 = min(x0 + 1, 127), y1 = min(y0 + 1, 127);
    const int i00 = y0 * W_ + x0, i01 = y0 * W_ + x1;
    const int i10 = y1 * W_ + x0, i11 = y1 * W_ + x1;
    const float w00 = (1.f - wx) * (1.f - wy), w01 = wx * (1.f - wy);
    const float w10 = (1.f - wx) * wy,         w11 = wx * wy;

    const float* xg = x + (size_t)(b * C_ + g * 32) * HW_;
    float* outp = up + (size_t)(b * C_ + g * 32) * OHW_ + oh * OW_ + ow;
    #pragma unroll 4
    for (int cc = 0; cc < 32; ++cc) {
        const float* xc = xg + (size_t)cc * HW_;
        const float v = xc[i00] * w00 + xc[i01] * w01 + xc[i10] * w10 + xc[i11] * w11;
        outp[(size_t)cc * OHW_] = v;
    }
}

// ---------------------------------------------------------------------------
// K4: depthwise 3x3 (pad 1) + per-channel sum/sumsq for BN batch stats.
__global__ __launch_bounds__(256) void k4_dw(
    const float* __restrict__ up, const float* __restrict__ wdw,
    float* __restrict__ y, float* __restrict__ stats)
{
    const int ow = threadIdx.x;
    const int oh = blockIdx.x;
    const int bc = blockIdx.y;          // b*128 + c
    const int c = bc & 127;
    const float* ib = up + (size_t)bc * OHW_;
    float wk[9];
    #pragma unroll
    for (int t = 0; t < 9; ++t) wk[t] = wdw[c * 9 + t];

    float acc = 0.0f;
    #pragma unroll
    for (int ky = 0; ky < 3; ++ky) {
        const int hh = oh + ky - 1;
        if (hh < 0 || hh >= OH_) continue;
        #pragma unroll
        for (int kx = 0; kx < 3; ++kx) {
            const int wp = ow + kx - 1;
            if (wp < 0 || wp >= OW_) continue;
            acc += wk[ky * 3 + kx] * ib[hh * OW_ + wp];
        }
    }
    y[(size_t)bc * OHW_ + oh * OW_ + ow] = acc;

    float s = acc, ss = acc * acc;      // wave32 butterfly reduce
    #pragma unroll
    for (int d = 16; d > 0; d >>= 1) {
        s  += __shfl_xor(s,  d, 32);
        ss += __shfl_xor(ss, d, 32);
    }
    if ((threadIdx.x & 31) == 0) {
        atomicAdd(&stats[c], s);
        atomicAdd(&stats[128 + c], ss);
    }
}

// ---------------------------------------------------------------------------
// K4b: fold BN (population stats) into per-channel scale a and shift b.
__global__ void k4b_finalize(const float* __restrict__ stats,
                             const float* __restrict__ gamma,
                             const float* __restrict__ beta,
                             float* __restrict__ ab)
{
    const int c = threadIdx.x;          // 128 threads
    const float N = (float)B_ * (float)OHW_;
    const float mu  = stats[c] / N;
    const float var = stats[128 + c] / N - mu * mu;
    const float a = gamma[c] * rsqrtf(var + 1e-5f);
    ab[c] = a;
    ab[128 + c] = beta[c] - mu * a;
}

// ---------------------------------------------------------------------------
// K5: fused BN+ReLU+1x1 conv 128->64 WMMA GEMM. Each wave: 4 N-tiles of 64x16.
__global__ __launch_bounds__(256) void k5_pw(
    const float* __restrict__ y, const float* __restrict__ wpw,
    const float* __restrict__ bpw, const float* __restrict__ ab,
    float* __restrict__ dout)
{
    __shared__ _Float16 wl[CP_ * C_];   // 64x128 f16 (16 KB)
    __shared__ float aS[128], bS[128];
    const int tid = threadIdx.x;
    for (int i = tid; i < CP_ * C_; i += 256) wl[i] = (_Float16)wpw[i];
    if (tid < 128) { aS[tid] = ab[tid]; bS[tid] = ab[128 + tid]; }
    __syncthreads();

    const int wave = tid >> 5, lane = tid & 31;
    const int g = lane >> 4, ln = lane & 15;
    const int b = blockIdx.y;
    const float* yb = y + (size_t)b * C_ * OHW_;
    float* ob = dout + (size_t)b * CP_ * OHW_;

    for (int it = 0; it < 4; ++it) {
        const int p0 = ((blockIdx.x * 8 + wave) * 4 + it) * 16;
        v8f acc[4] = {};
        for (int kk = 0; kk < C_; kk += 32) {
            v16h bf;
            const int kb = kk + g * 16;
            #pragma unroll
            for (int j = 0; j < 16; ++j) {
                const int c = kb + j;
                float v = yb[(size_t)c * OHW_ + p0 + ln];
                v = fmaxf(aS[c] * v + bS[c], 0.0f);     // BN + ReLU fused
                bf[j] = (_Float16)v;
            }
            #pragma unroll
            for (int mt = 0; mt < 4; ++mt) {
                const int m = mt * 16 + ln;
                const int base = m * C_ + kk + g * 16;
                v8h lo = *(const v8h*)&wl[base];
                v8h hi = *(const v8h*)&wl[base + 8];
                v16h af;
                #pragma unroll
                for (int j = 0; j < 8; ++j) { af[j] = lo[j]; af[j + 8] = hi[j]; }
                acc[mt] = WMMA_F16(af, bf, acc[mt]);
            }
        }
        #pragma unroll
        for (int mt = 0; mt < 4; ++mt) {
            #pragma unroll
            for (int v = 0; v < 8; ++v) {
                const int m = mt * 16 + v + g * 8;
                ob[(size_t)m * OHW_ + p0 + ln] = acc[mt][v] + bpw[m];
            }
        }
    }
}

// ---------------------------------------------------------------------------
extern "C" void kernel_launch(void* const* d_in, const int* in_sizes, int n_in,
                              void* d_out, int out_size, void* d_ws, size_t ws_size,
                              hipStream_t stream)
{
    const float* x      = (const float*)d_in[0];
    const float* w_off1 = (const float*)d_in[1];
    const float* b_off1 = (const float*)d_in[2];
    const float* w_off2 = (const float*)d_in[3];
    const float* w_dw   = (const float*)d_in[4];
    const float* gamma  = (const float*)d_in[5];
    const float* beta   = (const float*)d_in[6];
    const float* w_pw   = (const float*)d_in[7];
    const float* b_pw   = (const float*)d_in[8];
    float* out = (float*)d_out;

    float* ws = (float*)d_ws;
    const size_t n_off = (size_t)B_ * CO_ * HW_;     // 4.19M floats
    const size_t n_big = (size_t)B_ * C_ * OHW_;     // 67.1M floats
    float* off1 = ws;
    float* off  = off1 + n_off;
    float* up   = off  + n_off;
    float* yb   = up   + n_big;
    float* st   = yb   + n_big;
    float* ab   = st   + 256;

    k0_zero    <<<1, 256, 0, stream>>>(st);
    k1_conv1x1 <<<dim3(128, 8),   256, 0, stream>>>(x, w_off1, b_off1, off1);
    k2_conv3x3d2<<<dim3(128, 8),  256, 0, stream>>>(off1, w_off2, off);
    k3_sample  <<<dim3(256, 32),  256, 0, stream>>>(x, off, up);
    k4_dw      <<<dim3(256, 1024),256, 0, stream>>>(up, w_dw, yb, st);
    k4b_finalize<<<1, 128, 0, stream>>>(st, gamma, beta, ab);
    k5_pw      <<<dim3(128, 8),   256, 0, stream>>>(yb, w_pw, b_pw, ab, out);
}
